// MoE_26250840113791
// MI455X (gfx1250) — compile-verified, matrix operand order checked by reading, not run on hardware
//
#include <hip/hip_runtime.h>
#include <math.h>

// ---------------- problem constants ----------------
#define B_    32
#define CIN   128
#define LEN   1024
#define NE    4
#define COUT  128
#define KW    64
#define LP    961            // LEN - KW + 1
#define KTOT  (CIN * KW)     // 8192
#define NT    128            // l-tile per workgroup
#define LTILES 8             // ceil(961/128)
#define YCNT  (B_ * 2 * LP)  // 61504

typedef __attribute__((ext_vector_type(16))) __bf16 v16bf;
typedef __attribute__((ext_vector_type(2)))  __bf16 bf16x2;
typedef __attribute__((ext_vector_type(8)))  float  v8f;
typedef __attribute__((ext_vector_type(2)))  float  f32x2;
typedef f32x2 __attribute__((aligned(4)))    f32x2_u;   // allow 4B-aligned b64 loads

static __device__ __forceinline__ unsigned int pack_bf16v(f32x2 v) {
    union { bf16x2 h; unsigned int u; } cv;
    cv.h = __builtin_convertvector(v, bf16x2);   // single v_cvt_pk_bf16_f32
    return cv.u;
}

// ---------------- kernel 1: gate_x = mean over L ----------------
__global__ void k_rowmean(const float* __restrict__ x, float* __restrict__ gate_x) {
    int row = blockIdx.x;                     // b*128 + c
    const float* xr = x + (size_t)row * LEN;
    int tid = threadIdx.x;
    float s = xr[tid] + xr[tid + 256] + xr[tid + 512] + xr[tid + 768];
    __shared__ float red[256];
    red[tid] = s;
    __syncthreads();
    for (int off = 128; off > 0; off >>= 1) {
        if (tid < off) red[tid] += red[tid + off];
        __syncthreads();
    }
    if (tid == 0) gate_x[row] = red[0] * (1.0f / 1024.0f);
}

// ---------------- kernel 2: gating + aux loss (one wave) ----------------
static __device__ __forceinline__ float cv2_4(const float* v) {
    float m = (v[0] + v[1] + v[2] + v[3]) * 0.25f;
    float s = 0.f;
    for (int i = 0; i < 4; i++) { float d = v[i] - m; s += d * d; }
    float var = s * (1.0f / 3.0f);            // ddof=1
    return var / (m * m + 1e-10f);
}

__global__ void k_gate(const float* __restrict__ gate_x, const float* __restrict__ noise,
                       const float* __restrict__ w_gate, const float* __restrict__ w_noise,
                       float* __restrict__ gates, float* __restrict__ loss_out) {
    int b = threadIdx.x;                      // 0..31, one lane per batch row
    float clean[NE], rawn[NE];
    for (int e = 0; e < NE; e++) { clean[e] = 0.f; rawn[e] = 0.f; }
    for (int c = 0; c < CIN; c++) {
        float g = gate_x[b * CIN + c];
        for (int e = 0; e < NE; e++) {
            clean[e] += g * w_gate[c * NE + e];
            rawn[e]  += g * w_noise[c * NE + e];
        }
    }
    float nstd[NE], noisy[NE];
    for (int e = 0; e < NE; e++) {
        float r  = rawn[e];
        float sp = (r > 20.f) ? r : log1pf(expf(r));
        nstd[e]  = sp + 0.01f;                            // NOISE_EPS
        noisy[e] = clean[e] + noise[b * NE + e] * nstd[e];
    }
    // softmax over experts (TEMP = 1)
    float mx = noisy[0];
    for (int e = 1; e < NE; e++) mx = fmaxf(mx, noisy[e]);
    float ex[NE], sum = 0.f;
    for (int e = 0; e < NE; e++) { ex[e] = expf(noisy[e] - mx); sum += ex[e]; }
    float probs[NE];
    for (int e = 0; e < NE; e++) probs[e] = ex[e] / sum;
    // top-3 of 4 (strict > keeps earliest index on ties, like lax.top_k)
    int i0 = 0;
    for (int i = 1; i < NE; i++) if (probs[i] > probs[i0]) i0 = i;
    int i1 = -1;
    for (int i = 0; i < NE; i++) if (i != i0 && (i1 < 0 || probs[i] > probs[i1])) i1 = i;
    int i2 = -1;
    for (int i = 0; i < NE; i++) if (i != i0 && i != i1 && (i2 < 0 || probs[i] > probs[i2])) i2 = i;
    float tv0 = probs[i0], tv1 = probs[i1], tv2 = probs[i2];
    // softmax over top-2 probabilities
    float m2 = fmaxf(tv0, tv1);
    float a0 = expf(tv0 - m2), a1 = expf(tv1 - m2);
    float inv = 1.f / (a0 + a1);
    float grow[NE] = {0.f, 0.f, 0.f, 0.f};
    grow[i0] = a0 * inv;
    grow[i1] = a1 * inv;
    for (int e = 0; e < NE; e++) gates[b * NE + e] = grow[e];
    // load estimate (replicates reference exactly, incl. prob-space thresholds)
    float thr_in = tv2, thr_out = tv1;
    const float inv_sqrt2 = 0.70710678118654752f;
    float loadc[NE];
    for (int e = 0; e < NE; e++) {
        bool  isin = noisy[e] > thr_in;
        float thr  = isin ? thr_in : thr_out;
        float z    = (clean[e] - thr) / nstd[e];
        loadc[e]   = 0.5f * (1.f + erff(z * inv_sqrt2));
    }
    // deterministic wave32 reductions over batch
    float impv[NE], ldv[NE];
    for (int e = 0; e < NE; e++) {
        float v = grow[e];
        for (int off = 16; off > 0; off >>= 1) v += __shfl_xor(v, off);
        impv[e] = v;
        float w = loadc[e];
        for (int off = 16; off > 0; off >>= 1) w += __shfl_xor(w, off);
        ldv[e] = w;
    }
    if (b == 0) loss_out[0] = 0.01f * (cv2_4(impv) + cv2_4(ldv));
}

// ---------------- kernel 3: W1 f32 -> bf16 (same [e][o][c*64+k] layout) ----------------
__global__ void k_w1bf(const float* __restrict__ W1, unsigned short* __restrict__ Wb) {
    size_t i = ((size_t)blockIdx.x * 256 + threadIdx.x) * 4;
    f32x2 a = *(const f32x2*)(W1 + i);
    f32x2 b = *(const f32x2*)(W1 + i + 2);
    *(uint2*)(Wb + i) = make_uint2(pack_bf16v(a), pack_bf16v(b));
}

// ---------------- kernel 4: implicit-GEMM conv + fused head ----------------
template <bool CLAMP>
__global__ __launch_bounds__(256) void k_conv(
    const float* __restrict__ x, const unsigned short* __restrict__ Wb,
    const float* __restrict__ gates, const float* __restrict__ b1,
    const float* __restrict__ W2, const float* __restrict__ b2,
    float* __restrict__ zws, int lt0)
{
    const int lt = lt0 + blockIdx.x;          // l-tile
    const int e  = blockIdx.y;                // 0..3  expert
    const int b  = blockIdx.z;                // 0..31 batch
    if (gates[b * NE + e] == 0.0f) return;    // skip unselected experts (uniform per block)

    const int l0   = lt * NT;
    const int tid  = threadIdx.x;
    const int lane = tid & 31;
    const int wv   = tid >> 5;                // 8 waves, wave wv owns M rows [16wv,16wv+16)
    const int nl   = lane & 15;

    __shared__ unsigned int pack[2][2][8 * 32 * 8]; // [c&1][kh] packed B frags (32KB)
    __shared__ float zpart[16][2][NT];              // deterministic cross-wave partials (16KB)

    // packed B-fragment word coordinates for this thread (8 words/thread/K-step)
    int p_off[8];
    for (int i = 0; i < 8; i++) {
        int w  = tid * 8 + i;
        int s  = w >> 8;                      // 0..7 subtile
        int ln = (w >> 3) & 31;
        int j  = w & 7;
        int kb = ((ln < 16) ? 0 : 8) + ((j < 4) ? 2 * j : 16 + 2 * (j - 4));
        int n  = s * 16 + (ln & 15);
        p_off[i] = n + kb;                    // add l0 + kh*32 at use
    }

    // A-fragment base: lane holds row m, K halves per ISA layout
    const int m = 16 * wv + nl;
    const unsigned short* aptr =
        Wb + ((size_t)(e * COUT + m)) * KTOT + ((lane < 16) ? 0 : 8);
    const float* xrow_base = x + ((size_t)b * CIN) * LEN;

    v8f acc[8];
#pragma unroll
    for (int s = 0; s < 8; s++) acc[s] = (v8f){0.f,0.f,0.f,0.f,0.f,0.f,0.f,0.f};

    for (int c = 0; c < CIN; c++) {
        const float* xr = xrow_base + (size_t)c * LEN;
        const int bufc = c & 1;
        __builtin_prefetch((const void*)(aptr + c * KW + KW), 0, 1);
        // build packed bf16 B fragments for BOTH K-halves (Toeplitz: x[b,c, l0+n+k])
#pragma unroll
        for (int kh = 0; kh < 2; kh++) {
            const int xbase = l0 + kh * 32;
            unsigned int pw[8];
#pragma unroll
            for (int i = 0; i < 8; i++) {
                int xpos = xbase + p_off[i];
                if (CLAMP) {
                    int x0 = (xpos     < LEN - 1) ? xpos     : LEN - 1;
                    int x1 = (xpos + 1 < LEN - 1) ? xpos + 1 : LEN - 1;
                    f32x2 v; v.x = xr[x0]; v.y = xr[x1];
                    pw[i] = pack_bf16v(v);
                } else {
                    f32x2 v = *(const f32x2_u*)(xr + xpos);   // one b64, 4B-aligned OK
                    pw[i] = pack_bf16v(v);
                }
            }
            uint4* dst = (uint4*)&pack[bufc][kh][tid * 8];
            dst[0] = make_uint4(pw[0], pw[1], pw[2], pw[3]);
            dst[1] = make_uint4(pw[4], pw[5], pw[6], pw[7]);
        }
        // A fragments for BOTH K-halves: one clause of 4x b128, overlaps barrier wait
        union AFu { uint4 q[2]; v16bf v; };
        AFu A0, A1;
        {
            const uint4* ap = (const uint4*)(aptr + c * KW);
            A0.q[0] = ap[0]; A0.q[1] = ap[2];   // kh=0: K = +0..7, +16..23
            A1.q[0] = ap[4]; A1.q[1] = ap[6];   // kh=1: K = +32..39, +48..55
        }
        __syncthreads();                      // one barrier per channel
        // consume both K-halves: 16 WMMAs.
        // Load all 8 B fragments of a half as one batch (distinct registers) so the
        // in-order DS returns allow partial s_wait_dscnt and WMMA/load overlap.
#pragma unroll
        for (int kh = 0; kh < 2; kh++) {
            const uint4* bp = (const uint4*)&pack[bufc][kh][0];
            const v16bf Av = kh ? A1.v : A0.v;
            union BFu { uint4 q[2]; v16bf v; } Bf[8];
#pragma unroll
            for (int s = 0; s < 8; s++) {
                Bf[s].q[0] = bp[s * 64 + lane * 2];
                Bf[s].q[1] = bp[s * 64 + lane * 2 + 1];
            }
#pragma unroll
            for (int s = 0; s < 8; s++) {
                acc[s] = __builtin_amdgcn_wmma_f32_16x16x32_bf16(
                    false, Av, false, Bf[s].v, (short)0, acc[s], false, false);
            }
        }
    }

    // ---- fused epilogue: relu(h+b1), 2-way head, deterministic reduce, softmax ----
    const int half  = (lane < 16) ? 0 : 1;
    const int obase = 16 * wv + half * 8;     // D layout: VGPR r -> M = r (+8 for hi lanes)
    float b1r[8], w20[8], w21[8];
#pragma unroll
    for (int r = 0; r < 8; r++) {
        int o  = obase + r;
        b1r[r] = b1[e * COUT + o];
        w20[r] = W2[(e * 2 + 0) * COUT + o];
        w21[r] = W2[(e * 2 + 1) * COUT + o];
    }
    const int slot = wv * 2 + half;
#pragma unroll
    for (int s = 0; s < 8; s++) {
        float pz0 = 0.f, pz1 = 0.f;
#pragma unroll
        for (int r = 0; r < 8; r++) {
            float h = acc[s][r] + b1r[r];
            h = fmaxf(h, 0.f);
            pz0 += h * w20[r];
            pz1 += h * w21[r];
        }
        zpart[slot][0][s * 16 + nl] = pz0;
        zpart[slot][1][s * 16 + nl] = pz1;
    }
    __syncthreads();
    if (tid < NT) {
        float s0 = b2[e * 2 + 0], s1 = b2[e * 2 + 1];
#pragma unroll
        for (int i = 0; i < 16; i++) { s0 += zpart[i][0][tid]; s1 += zpart[i][1][tid]; }
        int l = l0 + tid;
        if (l < LP) {
            float mx = fmaxf(s0, s1);
            float e0 = expf(s0 - mx), e1 = expf(s1 - mx);
            float inv = 1.f / (e0 + e1);
            size_t base = ((size_t)((b * NE + e) * 2)) * LP + l;
            zws[base]      = e0 * inv;
            zws[base + LP] = e1 * inv;
        }
    }
}

// ---------------- kernel 5: gate-weighted combine ----------------
__global__ void k_combine(const float* __restrict__ zws, const float* __restrict__ gates,
                          float* __restrict__ y) {
    int idx = blockIdx.x * 256 + threadIdx.x;
    if (idx >= YCNT) return;
    int b   = idx / (2 * LP);
    int rem = idx - b * 2 * LP;
    int t   = rem / LP;
    int l   = rem - t * LP;
    float s = 0.f;
    for (int e = 0; e < NE; e++) {
        float g = gates[b * NE + e];
        if (g != 0.f) s += g * zws[((size_t)((b * NE + e) * 2 + t)) * LP + l];
    }
    y[idx] = s;
}

// ---------------- launcher ----------------
extern "C" void kernel_launch(void* const* d_in, const int* in_sizes, int n_in,
                              void* d_out, int out_size, void* d_ws, size_t ws_size,
                              hipStream_t stream) {
    (void)in_sizes; (void)n_in; (void)out_size; (void)ws_size;
    const float* x       = (const float*)d_in[0];
    const float* noise   = (const float*)d_in[1];
    const float* w_gate  = (const float*)d_in[2];
    const float* w_noise = (const float*)d_in[3];
    const float* W1      = (const float*)d_in[4];
    const float* b1      = (const float*)d_in[5];
    const float* W2      = (const float*)d_in[6];
    const float* b2      = (const float*)d_in[7];
    float* y = (float*)d_out;

    // workspace layout (bytes): Wb(bf16) | gate_x | gates | zws   (~9.4 MB total)
    char* ws = (char*)d_ws;
    const size_t wbBytes = (size_t)NE * COUT * KTOT * 2;          // 8,388,608
    unsigned short* Wb   = (unsigned short*)ws;
    float* gate_x        = (float*)(ws + wbBytes);                // 32*128
    float* gates         = gate_x + B_ * CIN;                     // 32*4
    float* zws           = gates + B_ * NE;                       // 32*4*2*961

    k_rowmean<<<dim3(B_ * CIN), dim3(256), 0, stream>>>(x, gate_x);
    k_gate<<<dim3(1), dim3(32), 0, stream>>>(gate_x, noise, w_gate, w_noise,
                                             gates, y + YCNT);
    k_w1bf<<<dim3((NE * COUT * KTOT) / (256 * 4)), dim3(256), 0, stream>>>(W1, Wb);
    // fast tiles 0..6 never run off the end of an x row; tile 7 needs clamping
    k_conv<false><<<dim3(LTILES - 1, NE, B_), 256, 0, stream>>>(x, Wb, gates, b1, W2, b2, zws, 0);
    k_conv<true ><<<dim3(1,          NE, B_), 256, 0, stream>>>(x, Wb, gates, b1, W2, b2, zws, LTILES - 1);
    k_combine<<<dim3((YCNT + 255) / 256), 256, 0, stream>>>(zws, gates, y);
}
